// FullColumn_43396349558829
// MI455X (gfx1250) — matrix operand order repaired and checked by reading, not compiled, not run on hardware
//
#include <hip/hip_runtime.h>
#include <hip/hip_bf16.h>

typedef __attribute__((ext_vector_type(16))) _Float16 v16h;
typedef __attribute__((ext_vector_type(8)))  _Float16 v8h;
typedef __attribute__((ext_vector_type(8)))  float    v8f;

#define SYN     512
#define NEUR    64
#define OC      10
#define ON      640          // OC*NEUR
#define KSIZE   48
#define T_IN    500
#define T_OUT   549
#define T_PAD   560          // 35 tiles of 16
#define TP      608          // padded transposed time rows: tp = t + 48, t in [-48, 560)
#define NB      8
#define THETA   25.6f
#define FODEP   48
#define WINNERS 16

#define CS_BLK   128         // cs per LDS stage (4 stages, double buffered)
#define TL       64          // time columns staged (need 63)
#define XS_PITCH 136         // 272B rows: 16B aligned, bank-staggered

// ---------------------------------------------------------------------------
// Step 0: time-major zero-padded f16 transpose of the spikes:
//   xT[b][tp][cs] = x[b][cs][tp-48]  (0 outside [0,500))
// Makes the conv's LDS tile a pure contiguous byte copy -> async-to-LDS path.
// ---------------------------------------------------------------------------
__global__ void transpose_x(const float* __restrict__ x,
                            _Float16* __restrict__ xT)
{
    int idx = blockIdx.x * blockDim.x + threadIdx.x;     // over NB*TP*SYN
    if (idx >= NB * TP * SYN) return;
    int cs = idx & (SYN - 1);
    int tp = (idx >> 9) % TP;
    int b  = idx / (TP * SYN);
    int t  = tp - KSIZE;
    float v = (t >= 0 && t < T_IN) ? x[((size_t)b * SYN + cs) * T_IN + t] : 0.0f;
    xT[idx] = (_Float16)v;
}

// ---------------------------------------------------------------------------
// Step 1: kern[j][on][cs] = max(0, min(j/16, 1.5*tanh(w) - j/32))  (unflipped;
// the conv uses x[t'-1-j] so the reference's flip is folded into the shift)
// ---------------------------------------------------------------------------
__global__ void build_kern(const float* __restrict__ weight,
                           _Float16* __restrict__ kern)
{
    int idx = blockIdx.x * blockDim.x + threadIdx.x;     // over ON*SYN
    if (idx >= ON * SYN) return;
    int on = idx / SYN, cs = idx % SYN;
    float w = tanhf(weight[idx]);
    #pragma unroll
    for (int j = 0; j < KSIZE; ++j) {
        float t_spike = (float)j * (1.0f / 16.0f);
        float t_leak  = 1.5f * w - (float)j * (1.0f / 32.0f);
        float v = fmaxf(0.0f, fminf(t_spike, t_leak));
        kern[((size_t)j * ON + on) * SYN + cs] = (_Float16)v;
    }
}

// ---------------------------------------------------------------------------
// Step 2: pot[b][on][t'] = sum_cs sum_j kern[j][on][cs] * x[b][cs][t'-1-j]
//                          + THETA*bias[on]
// WMMA f32_16x16x32_f16. Each wave owns TWO 16-row on-tiles sharing one
// B-fragment. x tile staged via async global->LDS B128 copies, double
// buffered over 4 cs-chunks, fenced by s_wait_asynccnt + barriers.
// ---------------------------------------------------------------------------
__global__ __launch_bounds__(128)
void conv_wmma(const _Float16* __restrict__ xT,    // [NB][TP][SYN]
               const _Float16* __restrict__ kern,  // [KSIZE][ON][SYN]
               const float* __restrict__ bias,     // [ON]
               float* __restrict__ pot)            // [NB][ON][T_PAD]
{
    __shared__ _Float16 xs[2][TL][XS_PITCH];

    const int tid  = threadIdx.x;
    const int wave = tid >> 5;
    const int lane = tid & 31;
    const int on_base = ((blockIdx.x * 4 + wave) * 2) * 16;  // two tiles: +0,+16
    const int t_base  = blockIdx.y * 16;                     // 35 t'-tiles
    const int b       = blockIdx.z;

    const int n  = lane & 15;    // A row (M) and B column (N)
    const int hl = lane >> 4;    // lane half selects K sub-runs

    // rows xT[b][t_base + tl][*] cover x times [t_base-48, t_base+15]
    const _Float16* xrow = xT + ((size_t)b * TP + t_base) * SYN;

    // async stage of one 64 x 128 f16 tile (1024 x 16B copies, 8 per thread)
    auto stage = [&](int buf, int csBlk) {
        #pragma unroll
        for (int i = 0; i < 8; ++i) {
            int idx = tid + i * 128;            // 0..1023
            int tl  = idx >> 4;                 // 0..63
            int ch  = idx & 15;                 // 16B chunk in row
            unsigned long long ga = (unsigned long long)(uintptr_t)
                (xrow + (size_t)tl * SYN + csBlk * CS_BLK + ch * 8);
            unsigned lds = (unsigned)(uintptr_t)&xs[buf][tl][ch * 8];
            asm volatile("global_load_async_to_lds_b128 %0, %1, off"
                         :: "v"(lds), "v"(ga) : "memory");
        }
    };

    stage(0, 0);
    asm volatile("s_wait_asynccnt 0x0" ::: "memory");
    __syncthreads();

    v8f c0 = {}, c1 = {};
    for (int csBlk = 0; csBlk < SYN / CS_BLK; ++csBlk) {
        const int buf = csBlk & 1;
        if (csBlk + 1 < SYN / CS_BLK) stage(buf ^ 1, csBlk + 1);

        for (int j = 0; j < KSIZE; ++j) {
            const int tl = n + (KSIZE - 1) - j;         // x column for this lane
            if (j + 1 < KSIZE)                          // warm WGP$ for next shift
                __builtin_prefetch(kern + ((size_t)(j + 1) * ON + on_base + n) * SYN
                                        + csBlk * CS_BLK, 0, 1);
            #pragma unroll
            for (int csc = 0; csc < CS_BLK / 32; ++csc) {
                // A (16-bit 16x32 ISA layout): lanes 0-15 K={0..7,16..23},
                // lanes 16-31 K={8..15,24..31}
                union { v16h v; v8h h[2]; } a0, a1, bm;
                const _Float16* ap = kern
                    + ((size_t)j * ON + on_base + n) * SYN
                    + csBlk * CS_BLK + csc * 32 + hl * 8;
                a0.h[0] = *(const v8h*)(ap);
                a0.h[1] = *(const v8h*)(ap + 16);
                a1.h[0] = *(const v8h*)(ap + 16 * SYN);
                a1.h[1] = *(const v8h*)(ap + 16 * SYN + 16);

                // B (32x16): same K order as A: lanes 0-15 K={0..7,16..23}, lanes 16-31 K={8..15,24..31}
                const _Float16* bp = &xs[buf][tl][csc * 32 + hl * 8];
                bm.h[0] = *(const v8h*)(bp);
                bm.h[1] = *(const v8h*)(bp + 16);

                c0 = __builtin_amdgcn_wmma_f32_16x16x32_f16(
                         false, a0.v, false, bm.v, (short)0, c0, false, false);
                c1 = __builtin_amdgcn_wmma_f32_16x16x32_f16(
                         false, a1.v, false, bm.v, (short)0, c1, false, false);
            }
        }
        __syncthreads();                                    // done reading buf
        asm volatile("s_wait_asynccnt 0x0" ::: "memory");   // my copies landed
        __syncthreads();                                    // everyone's copies landed
    }

    // C/D layout: VGPR r -> M = r + 8*half, N = lane&15
    #pragma unroll
    for (int r = 0; r < 8; ++r) {
        int m = r + hl * 8;
        pot[((size_t)b * ON + on_base + m) * T_PAD + t_base + n] =
            c0[r] + THETA * bias[on_base + m];
        pot[((size_t)b * ON + on_base + 16 + m) * T_PAD + t_base + n] =
            c1[r] + THETA * bias[on_base + 16 + m];
    }
}

// ---------------------------------------------------------------------------
// Step 3: sequential k-WTA with forced depression. One block per batch.
// ---------------------------------------------------------------------------
__global__ __launch_bounds__(256)
void wta(const float* __restrict__ pot,   // [NB][ON][T_PAD]
         float* __restrict__ out)         // [NB][OC][NEUR][T_OUT]
{
    __shared__ float smax[256];
    __shared__ int   sidx[256];
    __shared__ int   dep[NEUR];
    __shared__ int   scnt;
    __shared__ int   swinner;

    const int tid = threadIdx.x;
    const int b   = blockIdx.x;

    float* outb = out + (size_t)b * OC * NEUR * T_OUT;
    for (int i = tid; i < OC * NEUR * T_OUT; i += 256) outb[i] = 0.0f;
    if (tid < NEUR) dep[tid] = 0;
    __syncthreads();

    for (int t = 0; t < T_OUT; ++t) {
        if (tid == 0) scnt = 0;
        __syncthreads();
        if (tid < NEUR && dep[tid] != 0) atomicAdd(&scnt, 1);
        __syncthreads();
        const bool k_ok = (scnt < WINNERS);

        float best = -1.0f;
        int   bidx = 0;
        for (int on = tid; on < ON; on += 256) {
            int nn  = on & (NEUR - 1);
            float v = pot[((size_t)b * ON + on) * T_PAD + t];
            v = (k_ok && dep[nn] == 0) ? v : 0.0f;
            if (v > best) { best = v; bidx = on; }
        }
        smax[tid] = best; sidx[tid] = bidx;
        __syncthreads();
        for (int s = 128; s > 0; s >>= 1) {
            if (tid < s && smax[tid + s] > smax[tid]) {
                smax[tid] = smax[tid + s];
                sidx[tid] = sidx[tid + s];
            }
            __syncthreads();
        }
        if (tid == 0) {
            swinner = (smax[0] > THETA) ? sidx[0] : -1;
            if (swinner >= 0) {
                int o  = swinner / NEUR;
                int nn = swinner & (NEUR - 1);
                outb[((size_t)o * NEUR + nn) * T_OUT + t] = 1.0f;
            }
        }
        __syncthreads();
        if (tid < NEUR) {
            int inc = (swinner >= 0 && (swinner & (NEUR - 1)) == tid) ? FODEP : 0;
            int d = dep[tid] + inc - 1;
            dep[tid] = min(max(d, 0), FODEP - 1);
        }
        __syncthreads();
    }
}

// ---------------------------------------------------------------------------
extern "C" void kernel_launch(void* const* d_in, const int* in_sizes, int n_in,
                              void* d_out, int out_size, void* d_ws, size_t ws_size,
                              hipStream_t stream)
{
    (void)in_sizes; (void)n_in; (void)out_size; (void)ws_size;
    const float* x      = (const float*)d_in[0];   // [8,1,512,500]
    const float* weight = (const float*)d_in[1];   // [640,512]
    const float* bias   = (const float*)d_in[2];   // [640]
    float* out          = (float*)d_out;           // [8,10,64,549]

    const size_t KERN_BYTES = (size_t)KSIZE * ON * SYN * sizeof(_Float16); // 31.5 MB
    const size_t XT_BYTES   = (size_t)NB * TP * SYN * sizeof(_Float16);    //  5.0 MB

    _Float16* kern = (_Float16*)d_ws;
    _Float16* xT   = (_Float16*)((char*)d_ws + KERN_BYTES);
    float*    pot  = (float*)((char*)d_ws + KERN_BYTES + XT_BYTES);        // 11.5 MB

    transpose_x<<<(NB * TP * SYN + 255) / 256, 256, 0, stream>>>(x, xT);
    build_kern<<<(ON * SYN + 255) / 256, 256, 0, stream>>>(weight, kern);
    conv_wmma<<<dim3(5, 35, NB), 128, 0, stream>>>(xT, kern, bias, pot);
    wta<<<NB, 256, 0, stream>>>(pot, out);
}
